// PatchSampleF_24283745091862
// MI455X (gfx1250) — matrix operand order repaired and checked by reading, not compiled, not run on hardware
//
#include <hip/hip_runtime.h>
#include <hip/hip_bf16.h>
#include <math.h>

typedef __attribute__((ext_vector_type(16))) _Float16 v16h;
typedef __attribute__((ext_vector_type(8)))  _Float16 v8h;
typedef __attribute__((ext_vector_type(8)))  float    v8f;

// ---------------- wave helpers (wave32) ----------------
__device__ __forceinline__ float wsum32(float v) {
#pragma unroll
    for (int o = 16; o > 0; o >>= 1) v += __shfl_down(v, o, 32);
    return v; // lane 0 holds full sum
}

// ---------------- WMMA fragment loaders ----------------
// A: 16x32 f16 tile, row-major source with leading dim lda (halves).
// lanes 0-15: row M=lane, K=0..15 (halves in K order); lanes 16-31: same rows, K=16..31.
__device__ __forceinline__ v16h load_a16(const _Float16* A, int lda, int k0, int lane) {
    const _Float16* p = A + (lane & 15) * lda + k0 + ((lane >> 4) << 4);
    return *(const v16h*)(p);
}

// B: 32x16 f16 tile where B[K][N] = W[N][K], W row-major with leading dim ldw.
// lanes 0-15: column N=lane, K=0..15 (halves in K order); lanes 16-31: K=16..31.
__device__ __forceinline__ v16h load_b16(const _Float16* Wt, int ldw, int k0, int lane) {
    const _Float16* p = Wt + (lane & 15) * ldw + k0 + ((lane >> 4) << 4);
    return *(const v16h*)(p);
}

// ---------------- fp32 -> f16 convert ----------------
__global__ void cvt_f16_kernel(const float* __restrict__ src, _Float16* __restrict__ dst, int n) {
    int i = blockIdx.x * 256 + threadIdx.x;
    if (i < n) dst[i] = (_Float16)src[i];
}

// ---------------- Branch 0: sample (top-4 of 8) + mean ----------------
// feat0 (8,512,8,8): element (b,c,pos) at b*512*64 + c*64 + pos
__global__ void sample0_kernel(const float* __restrict__ feat0,
                               const int* __restrict__ patch_id0,
                               const int* __restrict__ local_id0,
                               _Float16* __restrict__ x0h) {
    int bidx = blockIdx.x;              // 512 = B(8) * N(64)
    int b = bidx >> 6, n = bidx & 63;
    int t = threadIdx.x;                // 256 threads
    int wave = t >> 5, lane = t & 31;
    __shared__ int   ids[9];
    __shared__ float red[17][8];
    __shared__ float sim[8];
    __shared__ int   sel[4];
    if (t == 0) ids[0] = patch_id0[n];
    if (t < 8)  ids[1 + t] = local_id0[n * 8 + t];
    __syncthreads();
    const float* base = feat0 + (size_t)b * 512 * 64;

    float dot[8] = {0,0,0,0,0,0,0,0};
    float nn[8]  = {0,0,0,0,0,0,0,0};
    float cc = 0.f;
    for (int c = t; c < 512; c += 256) {
        float cv = base[c * 64 + ids[0]];
        cc += cv * cv;
#pragma unroll
        for (int i = 0; i < 8; ++i) {
            float v = base[c * 64 + ids[1 + i]];
            dot[i] += cv * v;
            nn[i]  += v * v;
        }
    }
#pragma unroll
    for (int i = 0; i < 8; ++i) {
        float v = wsum32(dot[i]);
        if (lane == 0) red[i][wave] = v;
        v = wsum32(nn[i]);
        if (lane == 0) red[8 + i][wave] = v;
    }
    {
        float v = wsum32(cc);
        if (lane == 0) red[16][wave] = v;
    }
    __syncthreads();
    if (t < 8) {
        float d = 0.f, q = 0.f, c2 = 0.f;
        for (int w = 0; w < 8; ++w) { d += red[t][w]; q += red[8 + t][w]; c2 += red[16][w]; }
        float denom = fmaxf(sqrtf(q), 1e-12f) * fmaxf(sqrtf(c2), 1e-12f);
        sim[t] = d / denom;
    }
    __syncthreads();
    if (t < 8) {
        int rank = 0;
        float si = sim[t];
        for (int j = 0; j < 8; ++j) {
            float sj = sim[j];
            if (sj > si || (sj == si && j < t)) rank++;
        }
        if (rank < 4) sel[rank] = t;   // descending sim, tie -> lower index (matches top_k)
    }
    __syncthreads();
    for (int c = t; c < 512; c += 256) {
        float s = base[c * 64 + ids[0]];
#pragma unroll
        for (int k = 0; k < 4; ++k) s += base[c * 64 + ids[1 + sel[k]]];
        x0h[(size_t)bidx * 512 + c] = (_Float16)(s / 5.0f);
    }
}

// ---------------- Branch 0: h = x0 @ W1^T + b1 (512x1024, K=512) ----------------
__global__ void gemm_h_kernel(const _Float16* __restrict__ x0h,
                              const _Float16* __restrict__ W1h,
                              const float* __restrict__ b1,
                              float* __restrict__ h) {
    int wave = threadIdx.x >> 5, lane = threadIdx.x & 31;
    int rowTile = blockIdx.x >> 3;                  // 32 row tiles
    int colTile = ((blockIdx.x & 7) << 3) + wave;   // 64 col tiles
    const _Float16* A  = x0h + (size_t)rowTile * 16 * 512;
    const _Float16* Wt = W1h + (size_t)colTile * 16 * 512;
    v8f c = {};
    for (int k0 = 0; k0 < 512; k0 += 32) {
        v16h a = load_a16(A, 512, k0, lane);
        v16h b = load_b16(Wt, 512, k0, lane);
        c = __builtin_amdgcn_wmma_f32_16x16x32_f16(false, a, false, b, (short)0, c, false, false);
    }
    int col = colTile * 16 + (lane & 15);
    float bias = b1[col];
    int rbase = rowTile * 16 + ((lane >> 4) << 3);
#pragma unroll
    for (int r = 0; r < 8; ++r) h[(size_t)(rbase + r) * 1024 + col] = c[r] + bias;
}

// ---------------- BatchNorm stats per column (over 512 rows) ----------------
__global__ void bn_stats_kernel(const float* __restrict__ h,
                                const float* __restrict__ gamma,
                                const float* __restrict__ beta,
                                float* __restrict__ scale,
                                float* __restrict__ shift) {
    int j = blockIdx.x;                 // 1024 columns
    int t = threadIdx.x, wave = t >> 5, lane = t & 31;
    __shared__ float rs[8], rq[8];
    float s = 0.f, q = 0.f;
    for (int i = t; i < 512; i += 256) {
        float v = h[(size_t)i * 1024 + j];
        s += v; q += v * v;
    }
    s = wsum32(s); q = wsum32(q);
    if (lane == 0) { rs[wave] = s; rq[wave] = q; }
    __syncthreads();
    if (t == 0) {
        float S = 0.f, Q = 0.f;
        for (int w = 0; w < 8; ++w) { S += rs[w]; Q += rq[w]; }
        float mu  = S / 512.f;
        float var = Q / 512.f - mu * mu;
        float sc  = gamma[j] * rsqrtf(var + 1e-5f);
        scale[j] = sc;
        shift[j] = beta[j] - mu * sc;
    }
}

// ---------------- BN apply + ReLU -> f16 ----------------
__global__ void bn_apply_kernel(const float* __restrict__ h,
                                const float* __restrict__ scale,
                                const float* __restrict__ shift,
                                _Float16* __restrict__ A2h) {
    int idx = blockIdx.x * 256 + threadIdx.x;   // 512*1024
    int j = idx & 1023;
    float v = h[idx] * scale[j] + shift[j];
    A2h[idx] = (_Float16)(v > 0.f ? v : 0.f);
}

// ---------------- out0 = A2 @ W2^T + b2 (512x256, K=1024) ----------------
__global__ void gemm_out0_kernel(const _Float16* __restrict__ A2h,
                                 const _Float16* __restrict__ W2h,
                                 const float* __restrict__ b2,
                                 float* __restrict__ out0) {
    int wave = threadIdx.x >> 5, lane = threadIdx.x & 31;
    int rowTile = blockIdx.x >> 1;                 // 32 row tiles
    int colTile = ((blockIdx.x & 1) << 3) + wave;  // 16 col tiles
    const _Float16* A  = A2h + (size_t)rowTile * 16 * 1024;
    const _Float16* Wt = W2h + (size_t)colTile * 16 * 1024;
    v8f c = {};
    for (int k0 = 0; k0 < 1024; k0 += 32) {
        v16h a = load_a16(A, 1024, k0, lane);
        v16h b = load_b16(Wt, 1024, k0, lane);
        c = __builtin_amdgcn_wmma_f32_16x16x32_f16(false, a, false, b, (short)0, c, false, false);
    }
    int col = colTile * 16 + (lane & 15);
    float bias = b2[col];
    int rbase = rowTile * 16 + ((lane >> 4) << 3);
#pragma unroll
    for (int r = 0; r < 8; ++r) out0[(size_t)(rbase + r) * 256 + col] = c[r] + bias;
}

// ---------------- Branch 1: sample (top-40 of 80) -> x1 tiles (41x64 f16) ----------------
// feat1 (8,64,128,128): element (b,c,pos) at b*64*16384 + c*16384 + pos
__global__ void sample1_kernel(const float* __restrict__ feat1,
                               const int* __restrict__ patch_id1,
                               const int* __restrict__ local_id1,
                               _Float16* __restrict__ x1h) {
    int bidx = blockIdx.x;               // 4096 = B(8) * N(512)
    int b = bidx >> 9, n = bidx & 511;
    int t = threadIdx.x, wave = t >> 5, lane = t & 31;
    __shared__ int   nid[80];
    __shared__ float cen[64];
    __shared__ float sim[80];
    __shared__ int   sel[40];
    __shared__ float ccinv;
    int center = patch_id1[n];
    if (t < 80) nid[t] = local_id1[n * 80 + t];
    const float* base = feat1 + (size_t)b * 64 * 16384;
    if (t < 64) cen[t] = base[(size_t)t * 16384 + center];
    __syncthreads();
    if (wave == 0) {
        float s = cen[lane] * cen[lane] + cen[lane + 32] * cen[lane + 32];
        s = wsum32(s);
        if (lane == 0) ccinv = 1.f / fmaxf(sqrtf(s), 1e-12f);
    }
    __syncthreads();
    for (int r = wave; r < 80; r += 8) {
        int id = nid[r];
        float v0 = base[(size_t)lane * 16384 + id];
        float v1 = base[(size_t)(lane + 32) * 16384 + id];
        float d = cen[lane] * v0 + cen[lane + 32] * v1;
        float q = v0 * v0 + v1 * v1;
        d = wsum32(d);
        q = wsum32(q);
        if (lane == 0) sim[r] = d / fmaxf(sqrtf(q), 1e-12f) * ccinv;
    }
    __syncthreads();
    if (t < 80) {
        int rank = 0;
        float si = sim[t];
        for (int j = 0; j < 80; ++j) {
            float sj = sim[j];
            if (sj > si || (sj == si && j < t)) rank++;
        }
        if (rank < 40) sel[rank] = t;
    }
    __syncthreads();
    _Float16* dst = x1h + (size_t)bidx * 41 * 64;
    for (int e = t; e < 41 * 64; e += 256) {
        int row = e >> 6, ch = e & 63;
        int src = (row == 0) ? center : nid[sel[row - 1]];
        dst[e] = (_Float16)base[(size_t)ch * 16384 + src];
    }
}

// ---------------- Branch 1 fused MLP: relu(X@W3^T+b3)@W4^T+b4, row-normalize ----------------
// 16 rows per block; layer-1 result staged in LDS (f16), layer-2 in LDS (f32).
__global__ void mlp1_kernel(const _Float16* __restrict__ x1h,
                            const _Float16* __restrict__ W3h, const float* __restrict__ b3,
                            const _Float16* __restrict__ W4h, const float* __restrict__ b4,
                            float* __restrict__ out1) {
    __shared__ __align__(32) _Float16 y1[16 * 256];
    __shared__ __align__(32) float    y2[16 * 256];
    int wave = threadIdx.x >> 5, lane = threadIdx.x & 31;
    size_t row0 = (size_t)blockIdx.x * 16;           // 10496 blocks * 16 = 167936 rows
    const _Float16* A = x1h + row0 * 64;

    // layer 1: K=64 -> 256 hidden; wave handles 2 of 16 column tiles
#pragma unroll
    for (int tt = 0; tt < 2; ++tt) {
        int colTile = wave + tt * 8;
        const _Float16* Wt = W3h + (size_t)colTile * 16 * 64;
        v8f c = {};
#pragma unroll
        for (int k0 = 0; k0 < 64; k0 += 32) {
            v16h a = load_a16(A, 64, k0, lane);
            v16h b = load_b16(Wt, 64, k0, lane);
            c = __builtin_amdgcn_wmma_f32_16x16x32_f16(false, a, false, b, (short)0, c, false, false);
        }
        int col = colTile * 16 + (lane & 15);
        float bias = b3[col];
        int rbase = (lane >> 4) << 3;
#pragma unroll
        for (int r = 0; r < 8; ++r) {
            float v = c[r] + bias;
            y1[(rbase + r) * 256 + col] = (_Float16)(v > 0.f ? v : 0.f);
        }
    }
    __syncthreads();

    // layer 2: K=256 -> 256 out; A comes from LDS
#pragma unroll
    for (int tt = 0; tt < 2; ++tt) {
        int colTile = wave + tt * 8;
        const _Float16* Wt = W4h + (size_t)colTile * 16 * 256;
        v8f c = {};
#pragma unroll
        for (int k0 = 0; k0 < 256; k0 += 32) {
            v16h a = load_a16((const _Float16*)y1, 256, k0, lane);
            v16h b = load_b16(Wt, 256, k0, lane);
            c = __builtin_amdgcn_wmma_f32_16x16x32_f16(false, a, false, b, (short)0, c, false, false);
        }
        int col = colTile * 16 + (lane & 15);
        float bias = b4[col];
        int rbase = (lane >> 4) << 3;
#pragma unroll
        for (int r = 0; r < 8; ++r) y2[(rbase + r) * 256 + col] = c[r] + bias;
    }
    __syncthreads();

    // row-wise L2 normalize: wave handles rows wave and wave+8
#pragma unroll
    for (int tt = 0; tt < 2; ++tt) {
        int row = wave + tt * 8;
        float ss = 0.f;
#pragma unroll
        for (int j = 0; j < 8; ++j) {
            float v = y2[row * 256 + lane + j * 32];
            ss += v * v;
        }
        ss = wsum32(ss);
        ss = __shfl(ss, 0, 32);
        float inv = 1.f / (sqrtf(ss) + 1e-7f);
        float* dst = out1 + (row0 + row) * 256;
#pragma unroll
        for (int j = 0; j < 8; ++j) dst[lane + j * 32] = y2[row * 256 + lane + j * 32] * inv;
    }
}

// ---------------- launcher ----------------
extern "C" void kernel_launch(void* const* d_in, const int* in_sizes, int n_in,
                              void* d_out, int out_size, void* d_ws, size_t ws_size,
                              hipStream_t stream) {
    const float* feat0 = (const float*)d_in[0];
    const float* feat1 = (const float*)d_in[1];
    const float* W1    = (const float*)d_in[2];
    const float* b1    = (const float*)d_in[3];
    const float* gamma = (const float*)d_in[4];
    const float* beta  = (const float*)d_in[5];
    const float* W2    = (const float*)d_in[6];
    const float* b2    = (const float*)d_in[7];
    const float* W3    = (const float*)d_in[8];
    const float* b3    = (const float*)d_in[9];
    const float* W4    = (const float*)d_in[10];
    const float* b4    = (const float*)d_in[11];
    const int* patch_id0 = (const int*)d_in[12];
    const int* patch_id1 = (const int*)d_in[13];
    const int* local_id0 = (const int*)d_in[14];
    const int* local_id1 = (const int*)d_in[15];

    char* ws = (char*)d_ws;
    size_t off = 0;
    auto alloc = [&](size_t bytes) -> void* {
        void* p = ws + off;
        off = (off + bytes + 255) & ~(size_t)255;
        return p;
    };
    _Float16* x0h  = (_Float16*)alloc((size_t)512 * 512 * 2);
    _Float16* W1h  = (_Float16*)alloc((size_t)1024 * 512 * 2);
    _Float16* W2h  = (_Float16*)alloc((size_t)256 * 1024 * 2);
    _Float16* W3h  = (_Float16*)alloc((size_t)256 * 64 * 2);
    _Float16* W4h  = (_Float16*)alloc((size_t)256 * 256 * 2);
    float*    h    = (float*)   alloc((size_t)512 * 1024 * 4);
    float*    scal = (float*)   alloc((size_t)1024 * 4);
    float*    shft = (float*)   alloc((size_t)1024 * 4);
    _Float16* A2h  = (_Float16*)alloc((size_t)512 * 1024 * 2);
    _Float16* x1h  = (_Float16*)alloc((size_t)167936 * 64 * 2);

    float* out0 = (float*)d_out;                   // 512*256
    float* out1 = out0 + (size_t)512 * 256;        // 167936*256

    // weight conversion to f16
    cvt_f16_kernel<<<(1024 * 512 + 255) / 256, 256, 0, stream>>>(W1, W1h, 1024 * 512);
    cvt_f16_kernel<<<(256 * 1024 + 255) / 256, 256, 0, stream>>>(W2, W2h, 256 * 1024);
    cvt_f16_kernel<<<(256 * 64 + 255) / 256,   256, 0, stream>>>(W3, W3h, 256 * 64);
    cvt_f16_kernel<<<(256 * 256 + 255) / 256,  256, 0, stream>>>(W4, W4h, 256 * 256);

    // branch 0
    sample0_kernel<<<512, 256, 0, stream>>>(feat0, patch_id0, local_id0, x0h);
    gemm_h_kernel<<<256, 256, 0, stream>>>(x0h, W1h, b1, h);
    bn_stats_kernel<<<1024, 256, 0, stream>>>(h, gamma, beta, scal, shft);
    bn_apply_kernel<<<2048, 256, 0, stream>>>(h, scal, shft, A2h);
    gemm_out0_kernel<<<64, 256, 0, stream>>>(A2h, W2h, b2, out0);

    // branch 1
    sample1_kernel<<<4096, 256, 0, stream>>>(feat1, patch_id1, local_id1, x1h);
    mlp1_kernel<<<10496, 256, 0, stream>>>(x1h, W3h, b3, W4h, b4, out1);
}
